// KorniaSift_4733053960494
// MI455X (gfx1250) — compile-verified, matrix operand order checked by reading, not run on hardware
//
#include <hip/hip_runtime.h>
#include <hip/hip_bf16.h>
#include <math.h>

// ---------------------------------------------------------------------------
// Kornia SIFTDescriptor (patch_size=32) fused pipeline for MI455X / gfx1250.
//  Kernel 1: linearize constant pooling weights into WMMA B-fragment order.
//  Kernel 2: one 256-thread block (8 wave32) per patch:
//     bilinear patch gather -> grad/mag/ori -> soft angle binning (f16 LDS)
//     -> pooling as GEMM pooled[16x16] = ang[16x1024] @ Wt[1024x16]
//        via v_wmma_f32_16x16x32_f16, K split across the 8 waves
//     -> LDS partial reduction -> L2 norm, clip 0.2, L2 norm -> 128 f32 out.
// ---------------------------------------------------------------------------

typedef __attribute__((ext_vector_type(16))) _Float16 v16h;
typedef __attribute__((ext_vector_type(8)))  float    v8f;

#define PS        32
#define PATCH_PIX 1024     // 32*32
#define KSTEPS    32       // 1024 / 32 (K per WMMA)
#define WAVES     8
#define THREADS   256
#define TWO_PI    6.283185307179586f

// CDNA5 ISA §7.12.2: 16-bit A-matrix 16x32 K placement per VGPR v and
// lane-group g (lanes 0-15 -> g=0, lanes 16-31 -> g=1). Packed f16 pair base K:
__device__ __forceinline__ int kbase(int v, int g) {
    return ((v < 4) ? (2 * v) : (8 + 2 * v)) + g * 8;
}

// Triangular pooling weight: W[sb][pix], sb = oy*4+ox; conv k=12, stride 8, pad 3.
__device__ __forceinline__ float wfun(int sb, int pix) {
    int oy = sb >> 2, ox = sb & 3;
    int y = pix >> 5, x = pix & 31;
    int ky = y - oy * 8 + 3;
    int kx = x - ox * 8 + 3;
    if (ky < 0 || ky >= 12 || kx < 0 || kx >= 12) return 0.0f;
    float ay = 6.0f - fabsf((float)ky + 0.5f - 6.0f);
    float ax = 6.0f - fabsf((float)kx + 0.5f - 6.0f);
    return ay * ax * (1.0f / 36.0f);
}

// Pre-linearize Wt (1024x16) into WMMA B-fragment order:
//   wfrag[(kk*8 + v)*32 + lane] = pack_f16x2( B[K0][n], B[K0+1][n] )
//   with lane = g*16 + n, K0 = kbase(v,g), B[K][n] = W[n][kk*32 + K].
__global__ void sift_init_wfrag(unsigned int* __restrict__ wfrag) {
    for (int i = threadIdx.x; i < KSTEPS * 8 * 32; i += THREADS) {
        int lane = i & 31;
        int v    = (i >> 5) & 7;
        int kk   = i >> 8;
        int g = lane >> 4;
        int n = lane & 15;
        int K0 = kbase(v, g);
        union { unsigned u; _Float16 h[2]; } pk;
        pk.h[0] = (_Float16)wfun(n, kk * 32 + K0);
        pk.h[1] = (_Float16)wfun(n, kk * 32 + K0 + 1);
        wfrag[i] = pk.u;
    }
}

__device__ __forceinline__ float fetch(const float* __restrict__ im,
                                       int yi, int xi, int H, int W) {
    bool valid = (xi >= 0) & (xi < W) & (yi >= 0) & (yi < H);
    int yc = min(max(yi, 0), H - 1);
    int xc = min(max(xi, 0), W - 1);
    float v = im[(size_t)yc * W + xc];
    return valid ? v : 0.0f;
}

__global__ __launch_bounds__(THREADS)
void sift_patch_desc(const float* __restrict__ img,   // (B,1,H,W)
                     const float* __restrict__ laf,   // (B,N,2,3)
                     const unsigned int* __restrict__ wfrag,
                     float* __restrict__ out,         // (B*N,128)
                     int N, int H, int W) {
    const int p    = blockIdx.x;       // patch id in [0, B*N)
    const int t    = threadIdx.x;
    const int lane = t & 31;
    const int wave = t >> 5;

    __shared__ float    patch[PATCH_PIX];                  // 4 KB
    __shared__ _Float16 ang[8 * PATCH_PIX];                // 16 KB
    __shared__ float    graw[PS];
    __shared__ float    ginv;
    __shared__ float    partial[WAVES][16][16];            // 8 KB
    __shared__ float    red[128];

    // LAF affine (uniform per block; hits scalar cache)
    const float* Lf = laf + (size_t)p * 6;
    const float a00 = Lf[0], a01 = Lf[1], a02 = Lf[2];
    const float a10 = Lf[3], a11 = Lf[4], a12 = Lf[5];
    const float* im = img + (size_t)(p / N) * H * W;

    // 1D Gaussian, sigma = PS/sqrt(2); normalize by (sum)^2 for outer product
    if (t < PS) {
        float x = (float)t - (PS - 1) * 0.5f;
        float s = (float)PS / 1.41421356237f;
        graw[t] = expf(-(x * x) / (2.0f * s * s));
    }
    __syncthreads();
    if (t == 0) {
        float s = 0.0f;
        for (int i = 0; i < PS; ++i) s += graw[i];
        ginv = 1.0f / (s * s);
    }

    // ---- bilinear patch extraction (grid_sample, align_corners=False) ----
    for (int k = 0; k < 4; ++k) {
        int pix = t + k * THREADS;
        int r = pix >> 5, c = pix & 31;
        float u = (2.0f * (float)c + 1.0f) / (float)PS - 1.0f;
        float v = (2.0f * (float)r + 1.0f) / (float)PS - 1.0f;
        float px = a00 * u + a01 * v + a02;
        float py = a10 * u + a11 * v + a12;
        float xs = px - 0.5f, ys = py - 0.5f;
        float x0 = floorf(xs), y0 = floorf(ys);
        float wx = xs - x0,  wy = ys - y0;
        int xi = (int)x0, yi = (int)y0;
        float v00 = fetch(im, yi,     xi,     H, W);
        float v01 = fetch(im, yi,     xi + 1, H, W);
        float v10 = fetch(im, yi + 1, xi,     H, W);
        float v11 = fetch(im, yi + 1, xi + 1, H, W);
        patch[pix] = v00 * (1 - wx) * (1 - wy) + v01 * wx * (1 - wy)
                   + v10 * (1 - wx) * wy       + v11 * wx * wy;
    }
    __syncthreads();

    // ---- gradients, magnitude/orientation, soft angle binning ----
    for (int k = 0; k < 4; ++k) {
        int pix = t + k * THREADS;
        int r = pix >> 5, c = pix & 31;
        int cm = max(c - 1, 0), cp = min(c + 1, 31);
        int rm = max(r - 1, 0), rp = min(r + 1, 31);
        float gx = (patch[r * 32 + cp] - patch[r * 32 + cm]) * 0.5f;
        float gy = (patch[rp * 32 + c] - patch[rm * 32 + c]) * 0.5f;
        float mag = sqrtf(gx * gx + gy * gy + 1e-8f) * graw[r] * graw[c] * ginv;
        float ori = atan2f(gy, gx + 1e-8f) + TWO_PI;
        float ob  = ori * (8.0f / TWO_PI);
        float bf  = floorf(ob);
        float w1  = ob - bf;
        int b0 = ((int)bf) & 7;
        int b1 = (b0 + 1) & 7;
        float w0m = (1.0f - w1) * mag;
        float w1m = w1 * mag;
        #pragma unroll
        for (int a = 0; a < 8; ++a) {
            float val = (a == b0 ? w0m : 0.0f) + (a == b1 ? w1m : 0.0f);
            ang[a * PATCH_PIX + pix] = (_Float16)val;
        }
    }
    __syncthreads();

    // ---- pooling as GEMM via v_wmma_f32_16x16x32_f16 ----
    // pooled[M=16 (8 angle bins + 8 zero rows)][N=16 spatial bins], K=1024.
    // Wave w accumulates k-steps [4w, 4w+4).
    v8f acc = {0.f, 0.f, 0.f, 0.f, 0.f, 0.f, 0.f, 0.f};
    const int m = lane & 15;
    const int g = lane >> 4;
    const int m7 = m & 7;
    for (int kk = wave * 4; kk < wave * 4 + 4; ++kk) {
        union { v16h v; unsigned u[8]; } Af, Bf;
        int pix0 = kk * 32;
        #pragma unroll
        for (int r8 = 0; r8 < 8; ++r8) {
            int K0 = kbase(r8, g);
            // branch-free: always load, zero the padded rows (keeps EXEC full)
            unsigned av = *(const unsigned*)&ang[m7 * PATCH_PIX + pix0 + K0];
            Af.u[r8] = (m < 8) ? av : 0u;
            Bf.u[r8] = wfrag[(kk * 8 + r8) * 32 + lane];
        }
        acc = __builtin_amdgcn_wmma_f32_16x16x32_f16(
                  false, Af.v, false, Bf.v, (short)0, acc, false, false);
    }
    // C layout (§7.12.2): VGPR r -> M = r + 8*g, N = lane&15
    #pragma unroll
    for (int r8 = 0; r8 < 8; ++r8)
        partial[wave][r8 + 8 * g][m] = acc[r8];
    __syncthreads();

    // ---- cross-wave reduce + L2 norm -> clip 0.2 -> L2 norm ----
    float dval = 0.0f;
    if (t < 128) {
        int a = t >> 4, sb = t & 15;            // desc idx = a*16 + oy*4 + ox
        #pragma unroll
        for (int w = 0; w < WAVES; ++w) dval += partial[w][a][sb];
        red[t] = dval * dval;
    }
    __syncthreads();
    for (int s = 64; s > 0; s >>= 1) {
        if (t < s) red[t] += red[t + s];
        __syncthreads();
    }
    float n1 = fmaxf(sqrtf(red[0]), 1e-12f);
    __syncthreads();

    float cval = 0.0f;
    if (t < 128) {
        cval = fminf(fmaxf(dval / n1, 0.0f), 0.2f);
        red[t] = cval * cval;
    }
    __syncthreads();
    for (int s = 64; s > 0; s >>= 1) {
        if (t < s) red[t] += red[t + s];
        __syncthreads();
    }
    float n2 = fmaxf(sqrtf(red[0]), 1e-12f);
    if (t < 128) out[(size_t)p * 128 + t] = cval / n2;
}

extern "C" void kernel_launch(void* const* d_in, const int* in_sizes, int n_in,
                              void* d_out, int out_size, void* d_ws, size_t ws_size,
                              hipStream_t stream) {
    const float* img = (const float*)d_in[0];   // (B,1,2048,2048) f32
    const float* laf = (const float*)d_in[1];   // (B,N,2,3) f32
    float* out = (float*)d_out;                 // desc(B*N*128) ++ laf(B*N*6)

    const int H = 2048, W = 2048;
    const int B  = in_sizes[0] / (H * W);
    const int BN = in_sizes[1] / 6;             // total patches
    const int N  = BN / B;

    unsigned int* wfrag = (unsigned int*)d_ws;  // 32 KB of B-fragments

    sift_init_wfrag<<<1, THREADS, 0, stream>>>(wfrag);
    sift_patch_desc<<<BN, THREADS, 0, stream>>>(img, laf, wfrag, out, N, H, W);

    // laf pass-through (second tuple output), graph-capture-safe async D2D
    hipMemcpyAsync(out + (size_t)BN * 128, laf, (size_t)BN * 6 * sizeof(float),
                   hipMemcpyDeviceToDevice, stream);
}
